// HB_FC_MLP_36867999269165
// MI455X (gfx1250) — compile-verified, matrix operand order checked by reading, not run on hardware
//
#include <hip/hip_runtime.h>
#include <hip/hip_bf16.h>
#include <math.h>

// ---------------------------------------------------------------------------
// Problem constants (from reference)
// ---------------------------------------------------------------------------
#define BATCHN 4096
#define NU     8
#define NA     64
#define APC    8
#define P_POW  10.0f
#define P_NORM (P_POW / (float)APC)      // 1.25
#define K1P    1056                      // 1032 padded to multiple of 32
#define NH1    2048
#define NH2    2048
#define NH3    1024
#define NFPAD  144                       // 136 padded to multiple of 16 (C leading dim)
#define NFREAL 136
#define NFROWS 256                       // Wtf rows allocated (covers ragged last block)

typedef __attribute__((ext_vector_type(16))) __bf16 v16bf;
typedef __attribute__((ext_vector_type(8)))  float  v8f;

// ---------------------------------------------------------------------------
// small complex helpers
// ---------------------------------------------------------------------------
__device__ inline float2 conjc(float2 a){ return make_float2(a.x, -a.y); }
__device__ inline float2 cadd(float2 a, float2 b){ return make_float2(a.x+b.x, a.y+b.y); }
__device__ inline float2 csub(float2 a, float2 b){ return make_float2(a.x-b.x, a.y-b.y); }
__device__ inline float2 cmul(float2 a, float2 b){ return make_float2(a.x*b.x - a.y*b.y, a.x*b.y + a.y*b.x); }
__device__ inline float2 cmulc(float2 a, float2 b){ // a * conj(b)
  return make_float2(a.x*b.x + a.y*b.y, a.y*b.x - a.x*b.y);
}

// ---------------------------------------------------------------------------
// 0) pack MLP input: [Hre flat | Him flat | P_input | zero pad] -> bf16
// ---------------------------------------------------------------------------
__global__ void pack_input(const float* __restrict__ Hre, const float* __restrict__ Him,
                           __hip_bfloat16* __restrict__ X0) {
  size_t idx = (size_t)blockIdx.x * blockDim.x + threadIdx.x;
  if (idx >= (size_t)BATCHN * K1P) return;
  int b = (int)(idx / K1P), c = (int)(idx % K1P);
  float v;
  if      (c < 512)  v = Hre[(size_t)b * 512 + c];
  else if (c < 1024) v = Him[(size_t)b * 512 + (c - 512)];
  else if (c < 1032) v = P_POW / (float)(NU * APC);   // 0.15625
  else               v = 0.f;
  X0[idx] = __float2bfloat16(v);
}

// ---------------------------------------------------------------------------
// 1) weight transpose + bf16 convert:  W[K,N] fp32 -> Wt[Np,Kp] bf16 (zero pad)
// ---------------------------------------------------------------------------
__global__ void cvt_wT(const float* __restrict__ W, __hip_bfloat16* __restrict__ Wt,
                       int K, int N, int Kp, int Np) {
  size_t idx = (size_t)blockIdx.x * blockDim.x + threadIdx.x;
  if (idx >= (size_t)Np * Kp) return;
  int n = (int)(idx / Kp), k = (int)(idx % Kp);
  float v = (k < K && n < N) ? W[(size_t)k * N + n] : 0.f;
  Wt[idx] = __float2bfloat16(v);
}

// ---------------------------------------------------------------------------
// 2) WMMA GEMM:  C[M,N] = A[M,K](bf16,row) x Wt[Nrows,K](bf16,row)^T + bias
//
//  Block = 128 threads (4 waves), output tile 64M x 128N.
//  Wave computes 64x32 via TM=4 x TN=2 v_wmma_f32_16x16x32_bf16.
//  The block-shared A tile (64x32 bf16) is double-buffered in LDS, filled with
//  CDNA5 async copies (global_load_async_to_lds_b128, ASYNCcnt-tracked),
//  overlapped with WMMA compute.  LDS row stride 80B -> conflict-free
//  ds_load_b128 fragment reads (bank stride 20 mod 64).
// ---------------------------------------------------------------------------
#define TM 4
#define TN 2
#define A_STRIDE 80                     // bytes per 32-element bf16 row in LDS
#define A_TILE_BYTES (64 * A_STRIDE)

__device__ inline v16bf load_fragG(const __bf16* p) {   // global, contiguous K run
  v16bf f;
  *reinterpret_cast<float4*>(&f)       = *reinterpret_cast<const float4*>(p);       // K: g*8 ..
  *(reinterpret_cast<float4*>(&f) + 1) = *reinterpret_cast<const float4*>(p + 16);  // K: 16+g*8 ..
  return f;
}

__global__ __launch_bounds__(128) void gemm_wmma(
    const __bf16* __restrict__ A, const __bf16* __restrict__ Bt,
    const float* __restrict__ bias, int biasN,
    float* __restrict__ C, int M, int N, int K, int applyTanh)
{
  __shared__ char shA[2][A_TILE_BYTES];

  const int tid   = threadIdx.x;
  const int lane  = tid & 31;
  const int wave  = tid >> 5;
  const int m0    = blockIdx.y * 64;
  const int nBase = blockIdx.x * (16 * TN * 4) + wave * (16 * TN);
  const int r     = lane & 15;          // row (A frag) / col (B frag) within tile
  const int g     = lane >> 4;          // K-octet selector

  const __bf16* aSrc   = A + (size_t)m0 * K;
  const __bf16* bBase0 = Bt + (size_t)(nBase + r) * K + g * 8;
  const __bf16* bBase1 = Bt + (size_t)(nBase + 16 + r) * K + g * 8;

  // async-copy mapping: 256 x 16B chunks per 64x32 tile, 2 per thread
  const int row0 = tid >> 2,          kc0 = tid & 3;          // chunk tid
  const int row1 = (tid + 128) >> 2,  kc1 = tid & 3;          // chunk tid+128

  v8f acc[TM][TN] = {};

  const int nk = K / 32;
  // prologue: request k-step 0 into buffer 0
  {
    unsigned l0 = (unsigned)(uintptr_t)&shA[0][row0 * A_STRIDE + kc0 * 16];
    unsigned l1 = (unsigned)(uintptr_t)&shA[0][row1 * A_STRIDE + kc1 * 16];
    const void* g0 = (const void*)(aSrc + (size_t)row0 * K + kc0 * 8);
    const void* g1 = (const void*)(aSrc + (size_t)row1 * K + kc1 * 8);
    asm volatile("global_load_async_to_lds_b128 %0, %1, off" :: "v"(l0), "v"(g0) : "memory");
    asm volatile("global_load_async_to_lds_b128 %0, %1, off" :: "v"(l1), "v"(g1) : "memory");
  }

  int cur = 0;
  for (int kt = 0; kt < nk; ++kt) {
    asm volatile("s_wait_asynccnt 0x0" ::: "memory");   // my async writes done
    __syncthreads();                                    // everyone's writes visible,
                                                        // prior-iter LDS reads drained
    if (kt + 1 < nk) {                                  // prefetch next tile (other buffer)
      int k0n = (kt + 1) * 32;
      unsigned l0 = (unsigned)(uintptr_t)&shA[cur ^ 1][row0 * A_STRIDE + kc0 * 16];
      unsigned l1 = (unsigned)(uintptr_t)&shA[cur ^ 1][row1 * A_STRIDE + kc1 * 16];
      const void* g0 = (const void*)(aSrc + (size_t)row0 * K + k0n + kc0 * 8);
      const void* g1 = (const void*)(aSrc + (size_t)row1 * K + k0n + kc1 * 8);
      asm volatile("global_load_async_to_lds_b128 %0, %1, off" :: "v"(l0), "v"(g0) : "memory");
      asm volatile("global_load_async_to_lds_b128 %0, %1, off" :: "v"(l1), "v"(g1) : "memory");
    }

    const int k0 = kt * 32;
    v16bf bf0 = load_fragG(bBase0 + k0);
    v16bf bf1 = load_fragG(bBase1 + k0);
#pragma unroll
    for (int t = 0; t < TM; ++t) {
      const char* ap = &shA[cur][(t * 16 + r) * A_STRIDE + g * 16];
      v16bf af;
      *reinterpret_cast<float4*>(&af)       = *reinterpret_cast<const float4*>(ap);      // K: g*8..
      *(reinterpret_cast<float4*>(&af) + 1) = *reinterpret_cast<const float4*>(ap + 32); // K: 16+g*8..
      acc[t][0] = __builtin_amdgcn_wmma_f32_16x16x32_bf16(
                      false, af, false, bf0, (short)0, acc[t][0], false, false);
      acc[t][1] = __builtin_amdgcn_wmma_f32_16x16x32_bf16(
                      false, af, false, bf1, (short)0, acc[t][1], false, false);
    }
    cur ^= 1;
  }

  // C/D layout: lane covers column (lane&15), rows q + 8*(lane>>4)
#pragma unroll
  for (int j = 0; j < TN; ++j) {
    const int n = nBase + j * 16 + r;
    if (n < N) {
      const float bv = (n < biasN) ? bias[n] : 0.f;
#pragma unroll
      for (int t = 0; t < TM; ++t) {
#pragma unroll
        for (int q = 0; q < 8; ++q) {
          int m   = m0 + t * 16 + q + 8 * g;
          float v = acc[t][j][q] + bv;
          if (applyTanh) v = tanhf(v);
          C[(size_t)m * N + n] = v;
        }
      }
    }
  }
}

// ---------------------------------------------------------------------------
// 3) BatchNorm batch statistics: per-column mean & biased variance over M rows
// ---------------------------------------------------------------------------
__global__ __launch_bounds__(256) void bn_stats(const float* __restrict__ Y,
                                                float* __restrict__ stats, int M, int N) {
  int c    = blockIdx.x * 64 + (threadIdx.x & 63);
  int part = threadIdx.x >> 6;                  // 4-way row partition
  float s = 0.f, s2 = 0.f;
  for (int m = part; m < M; m += 4) {
    float v = Y[(size_t)m * N + c];
    s += v; s2 += v * v;
  }
  __shared__ float shs[4][64], shq[4][64];
  shs[part][threadIdx.x & 63] = s;
  shq[part][threadIdx.x & 63] = s2;
  __syncthreads();
  if (part == 0) {
    int l = threadIdx.x & 63;
    s  = shs[0][l] + shs[1][l] + shs[2][l] + shs[3][l];
    s2 = shq[0][l] + shq[1][l] + shq[2][l] + shq[3][l];
    float mean = s / (float)M;
    stats[c]     = mean;
    stats[N + c] = s2 / (float)M - mean * mean; // biased variance (ddof=0)
  }
}

// ---------------------------------------------------------------------------
// 4) BN normalize + ReLU + convert to bf16 activation
// ---------------------------------------------------------------------------
__global__ void bn_relu_cvt(const float* __restrict__ Y, const float* __restrict__ stats,
                            const float* __restrict__ gam, const float* __restrict__ bet,
                            __hip_bfloat16* __restrict__ X, int M, int N) {
  size_t idx = (size_t)blockIdx.x * blockDim.x + threadIdx.x;
  if (idx >= (size_t)M * N) return;
  int c = (int)(idx % N);
  float sc = gam[c] * rsqrtf(stats[N + c] + 1e-5f);
  float v  = (Y[idx] - stats[c]) * sc + bet[c];
  X[idx] = __float2bfloat16(fmaxf(v, 0.f));
}

// ---------------------------------------------------------------------------
// 5) Post-processing: F, V (RZF), Rate.  One 64-thread workgroup per batch elem.
// ---------------------------------------------------------------------------
__global__ __launch_bounds__(64) void postproc(
    const float* __restrict__ mlpOut,           // [B, NFPAD], tanh already applied
    const float* __restrict__ Hre, const float* __restrict__ Him,
    float* __restrict__ Fout, float* __restrict__ Vout, float* __restrict__ Rate)
{
  const int b = blockIdx.x;
  const int t = threadIdx.x;                    // 0..63
  __shared__ float2 Fu[64], Heff[64], Maug[8][16], Wm[64], facs[8];
  __shared__ float  T[64], P2[8], sums[8];

  const float* o = mlpOut + (size_t)b * NFPAD;

  // --- unit-modulus phases + block-diagonal F output ---
  {
    float re = o[t], im = o[64 + t];
    float inv = rsqrtf(re * re + im * im);
    float2 fu = make_float2(re * inv, im * inv);
    Fu[t] = fu;
    int rchain = t >> 3;                        // antenna t -> RF chain t/APC
    float2* Fo = reinterpret_cast<float2*>(Fout) + (size_t)b * (NA * 8) + (size_t)t * 8;
#pragma unroll
    for (int r = 0; r < 8; ++r) Fo[r] = (r == rchain) ? fu : make_float2(0.f, 0.f);
  }
  if (t < 8) { float p = 1.f / (1.f + expf(-o[128 + t])); P2[t] = p; }
  __syncthreads();

  if (t == 0) {                                 // power allocation P2 = P_NORM * p / sum(p)
    float s = 0.f;
    for (int u = 0; u < 8; ++u) s += P2[u];
    float c = P_NORM / s;
    for (int u = 0; u < 8; ++u) P2[u] *= c;
  }
  // --- Heff[u][r] = sum_j H[u][r*8+j] * Fu[r*8+j]  (block-diagonal F) ---
  {
    int u = t >> 3, r = t & 7;
    const float* hr = Hre + ((size_t)b * 8 + u) * 64 + r * 8;
    const float* hi = Him + ((size_t)b * 8 + u) * 64 + r * 8;
    float2 acc = make_float2(0.f, 0.f);
#pragma unroll
    for (int j = 0; j < 8; ++j) acc = cadd(acc, cmul(make_float2(hr[j], hi[j]), Fu[r * 8 + j]));
    Heff[t] = acc;
  }
  __syncthreads();

  // --- G = Heff Heff^H + (U/P_NORM) I ; augmented [G | I] ---
  {
    int u = t >> 3, v = t & 7;
    float2 g = make_float2(0.f, 0.f);
#pragma unroll
    for (int r = 0; r < 8; ++r) g = cadd(g, cmulc(Heff[u * 8 + r], Heff[v * 8 + r]));
    if (u == v) g.x += (float)NU / P_NORM;      // 6.4
    Maug[u][v]     = g;
    Maug[u][8 + v] = make_float2((u == v) ? 1.f : 0.f, 0.f);
  }
  __syncthreads();

  // --- Gauss-Jordan inverse of 8x8 complex G ---
  for (int k = 0; k < 8; ++k) {
    if (t == 0) {
      float2 p = Maug[k][k];
      float  d = 1.f / (p.x * p.x + p.y * p.y);
      facs[0] = make_float2(p.x * d, -p.y * d); // 1/pivot
    }
    __syncthreads();
    if (t < 16) Maug[k][t] = cmul(Maug[k][t], facs[0]);
    __syncthreads();
    if (t < 8) facs[t] = Maug[t][k];            // elimination factors (facs[k]==1, skipped)
    __syncthreads();
    for (int q = t; q < 128; q += 64) {
      int u = q >> 4, c = q & 15;
      if (u != k) Maug[u][c] = csub(Maug[u][c], cmul(facs[u], Maug[k][c]));
    }
    __syncthreads();
  }

  // --- Wm[r][u] = sum_w conj(Heff[w][r]) * Ginv[w][u] ---
  {
    int r = t >> 3, u = t & 7;
    float2 acc = make_float2(0.f, 0.f);
#pragma unroll
    for (int w = 0; w < 8; ++w) acc = cadd(acc, cmul(conjc(Heff[w * 8 + r]), Maug[w][8 + u]));
    Wm[r * 8 + u] = acc;
  }
  __syncthreads();

  if (t < 8) {                                  // per-user column scale: 1/||col|| * sqrt(P2)
    float s = 0.f;
    for (int r = 0; r < 8; ++r) { float2 w = Wm[r * 8 + t]; s += w.x * w.x + w.y * w.y; }
    sums[t] = rsqrtf(s) * sqrtf(P2[t]);
  }
  __syncthreads();

  {                                             // V = Wn * sqrt(P2); write V output
    int r = t >> 3, u = t & 7;
    float2 v = make_float2(Wm[r * 8 + u].x * sums[u], Wm[r * 8 + u].y * sums[u]);
    Wm[r * 8 + u] = v;
    reinterpret_cast<float2*>(Vout)[(size_t)b * 64 + r * 8 + u] = v;
  }
  __syncthreads();

  {                                             // T[u][v] = |(Heff V)[u][v]|^2
    int u = t >> 3, v = t & 7;
    float2 e = make_float2(0.f, 0.f);
#pragma unroll
    for (int r = 0; r < 8; ++r) e = cadd(e, cmul(Heff[u * 8 + r], Wm[r * 8 + v]));
    T[t] = e.x * e.x + e.y * e.y;
  }
  __syncthreads();

  if (t < 8) {
    float sig = T[t * 8 + t], tot = 0.f;
    for (int v = 0; v < 8; ++v) tot += T[t * 8 + v];
    sums[t] = log2f(1.f + sig / (tot - sig + 1.f));
  }
  __syncthreads();
  if (t == 0) {
    float r = 0.f;
    for (int u = 0; u < 8; ++u) r += sums[u];
    Rate[b] = r;
  }
}

// ---------------------------------------------------------------------------
// launch
// ---------------------------------------------------------------------------
extern "C" void kernel_launch(void* const* d_in, const int* in_sizes, int n_in,
                              void* d_out, int out_size, void* d_ws, size_t ws_size,
                              hipStream_t stream) {
  (void)in_sizes; (void)n_in; (void)out_size; (void)ws_size;
  const float* Hre = (const float*)d_in[0];
  const float* Him = (const float*)d_in[1];
  const float* W1  = (const float*)d_in[2];  const float* b1  = (const float*)d_in[3];
  const float* g1  = (const float*)d_in[4];  const float* be1 = (const float*)d_in[5];
  const float* W2  = (const float*)d_in[6];  const float* b2  = (const float*)d_in[7];
  const float* g2  = (const float*)d_in[8];  const float* be2 = (const float*)d_in[9];
  const float* W3  = (const float*)d_in[10]; const float* b3  = (const float*)d_in[11];
  const float* g3  = (const float*)d_in[12]; const float* be3 = (const float*)d_in[13];
  const float* Wf  = (const float*)d_in[14]; const float* bfb = (const float*)d_in[15];

  // workspace carve (deterministic)
  char* ws = (char*)d_ws;
  size_t off = 0;
  auto carve = [&](size_t bytes) -> char* {
    char* p = ws + off; off = (off + bytes + 255) & ~(size_t)255; return p;
  };
  __hip_bfloat16* X0  = (__hip_bfloat16*)carve((size_t)BATCHN * K1P * 2);   // also reused as X3
  __hip_bfloat16* X1  = (__hip_bfloat16*)carve((size_t)BATCHN * NH1 * 2);   // also reused as X2
  __hip_bfloat16* Wt1 = (__hip_bfloat16*)carve((size_t)NH1 * K1P * 2);
  __hip_bfloat16* Wt2 = (__hip_bfloat16*)carve((size_t)NH2 * NH1 * 2);
  __hip_bfloat16* Wt3 = (__hip_bfloat16*)carve((size_t)NH3 * NH2 * 2);
  __hip_bfloat16* Wtf = (__hip_bfloat16*)carve((size_t)NFROWS * NH3 * 2);   // 256 rows, zero-padded
  float*          Y   = (float*)carve((size_t)BATCHN * NH1 * 4);            // pre-BN / final fp32
  float*          st  = (float*)carve((size_t)2 * NH1 * 4);
  __hip_bfloat16* X2 = X1;
  __hip_bfloat16* X3 = X0;

  float* Fout = (float*)d_out;
  float* Vout = Fout + (size_t)BATCHN * NA * 8 * 2;   // after F (complex)
  float* Rate = Vout + (size_t)BATCHN * 8 * 8 * 2;    // after V (complex)

  // --- prep ---
  pack_input<<<((size_t)BATCHN * K1P + 255) / 256, 256, 0, stream>>>(Hre, Him, X0);
  cvt_wT<<<((size_t)NH1 * K1P + 255) / 256, 256, 0, stream>>>(W1, Wt1, 1032, NH1, K1P, NH1);
  cvt_wT<<<((size_t)NH2 * NH1 + 255) / 256, 256, 0, stream>>>(W2, Wt2, NH1, NH2, NH1, NH2);
  cvt_wT<<<((size_t)NH3 * NH2 + 255) / 256, 256, 0, stream>>>(W3, Wt3, NH2, NH3, NH2, NH3);
  cvt_wT<<<((size_t)NFROWS * NH3 + 255) / 256, 256, 0, stream>>>(Wf, Wtf, NH3, NFREAL, NH3, NFROWS);

  dim3 blk(128);
  // --- layer 1 ---
  gemm_wmma<<<dim3(NH1 / 128, BATCHN / 64), blk, 0, stream>>>(
      (const __bf16*)X0, (const __bf16*)Wt1, b1, NH1, Y, BATCHN, NH1, K1P, 0);
  bn_stats<<<NH1 / 64, 256, 0, stream>>>(Y, st, BATCHN, NH1);
  bn_relu_cvt<<<((size_t)BATCHN * NH1 + 255) / 256, 256, 0, stream>>>(Y, st, g1, be1, X1, BATCHN, NH1);
  // --- layer 2 ---
  gemm_wmma<<<dim3(NH2 / 128, BATCHN / 64), blk, 0, stream>>>(
      (const __bf16*)X1, (const __bf16*)Wt2, b2, NH2, Y, BATCHN, NH2, NH1, 0);
  bn_stats<<<NH2 / 64, 256, 0, stream>>>(Y, st, BATCHN, NH2);
  bn_relu_cvt<<<((size_t)BATCHN * NH2 + 255) / 256, 256, 0, stream>>>(Y, st, g2, be2, X2, BATCHN, NH2);
  // --- layer 3 ---
  gemm_wmma<<<dim3(NH3 / 128, BATCHN / 64), blk, 0, stream>>>(
      (const __bf16*)X2, (const __bf16*)Wt3, b3, NH3, Y, BATCHN, NH3, NH2, 0);
  bn_stats<<<NH3 / 64, 256, 0, stream>>>(Y, st, BATCHN, NH3);
  bn_relu_cvt<<<((size_t)BATCHN * NH3 + 255) / 256, 256, 0, stream>>>(Y, st, g3, be3, X3, BATCHN, NH3);
  // --- final layer (tanh) ---
  gemm_wmma<<<dim3((NFPAD + 127) / 128, BATCHN / 64), blk, 0, stream>>>(
      (const __bf16*)X3, (const __bf16*)Wtf, bfb, NFREAL, Y, BATCHN, NFPAD, NH3, 1);
  // --- beamforming post-processing ---
  postproc<<<BATCHN, 64, 0, stream>>>(Y, Hre, Him, Fout, Vout, Rate);
}
